// DIN_76484777607322
// MI455X (gfx1250) — matrix operand order repaired and checked by reading, not compiled, not run on hardware
//
#include <hip/hip_runtime.h>
#include <hip/hip_bf16.h>

typedef __attribute__((ext_vector_type(16))) __bf16 v16bf;
typedef __attribute__((ext_vector_type(8)))  __bf16 v8bf;
typedef __attribute__((ext_vector_type(8)))  float  v8f;

#define WMMA_BF16(A,Bm,C) \
  __builtin_amdgcn_wmma_f32_16x16x32_bf16(false,(A),false,(Bm),(short)0,(C),false,false)

#define B_SZ   4096
#define S_SEQ  200
#define SP     224     // padded sequence (14 * 16) -> 7 software-pipelined tile pairs
#define E_DIM  64
#define H_DIM  128
#define ROWP   72      // padded bf16 row stride (64 + 8) -> 144B, bank-friendly

// B-fragment (32x16, K x N): elems j<8: K = base+hf*8+j ; j>=8: K = base+16+hf*8+(j-8),
// N = lane&15 -> sequence row. Two 8-bf16 chunks from the behavior row.
__device__ __forceinline__ v16bf b_frag_beh(const __bf16* __restrict__ r, int ebase, int hf) {
  v8bf c1 = *(const v8bf*)(r + ebase + hf * 8);
  v8bf c2 = *(const v8bf*)(r + ebase + 16 + hf * 8);
  return __builtin_shufflevector(c1, c2, 0,1,2,3,4,5,6,7,8,9,10,11,12,13,14,15);
}

__global__ __launch_bounds__(256) void din_kernel(
    const int* __restrict__ uf, const int* __restrict__ tif, const int* __restrict__ seq,
    const float* __restrict__ eu0, const float* __restrict__ eu1, const float* __restrict__ ei,
    const float* __restrict__ aw1, const float* __restrict__ ab1,
    const float* __restrict__ aw2, const float* __restrict__ ab2,
    const float* __restrict__ fw1, const float* __restrict__ fb1,
    const float* __restrict__ fw2, const float* __restrict__ fb2,
    float* __restrict__ out)
{
  __shared__ __bf16 s_beh[SP * ROWP];   // ~31.5 KB behavior tile (bf16)
  __shared__ float  s_t[E_DIM];         // target (f32)
  __shared__ float  s_tvec[H_DIM];      // t@(W1b-W1c) + b1
  __shared__ float  s_scores[SP];
  __shared__ float  s_comb[256];        // [u0(64) | u1(64) | interest(64) | target(64)]
  __shared__ float  s_red[256];
  __shared__ int    s_sidx[SP];
  __shared__ float  s_inv;

  const int b    = blockIdx.x;
  const int tid  = threadIdx.x;
  const int lane = tid & 31;
  const int wv   = tid >> 5;    // wave id -> H (M) tile
  const int li   = lane & 15;
  const int hf   = lane >> 4;

  // ---------- phase 0: indices, target, user embeds, zero scores ----------
  if (tid < SP) {
    s_sidx[tid]   = (tid < S_SEQ) ? seq[b * S_SEQ + tid] : -1;
    s_scores[tid] = 0.f;
  }
  if (tid < 64) {
    float tv = ei[(long)tif[b] * E_DIM + tid];
    s_t[tid] = tv;
    s_comb[192 + tid] = tv;
  } else if (tid < 128) {
    s_comb[tid - 64] = eu0[(long)uf[2 * b] * E_DIM + (tid - 64)];
  } else if (tid < 192) {
    s_comb[tid - 64] = eu1[(long)uf[2 * b + 1] * E_DIM + (tid - 128)];
  }
  __syncthreads();

  // ---------- phase 0b: tvec + gather behavior rows (float4 -> bf16) ----------
  // tvec[h] = b1[h] + sum_e t[e] * (W1b[e][h] - W1c[e][h])
  if (tid < H_DIM) {
    float tb = ab1[tid];
#pragma unroll 8
    for (int e = 0; e < E_DIM; ++e)
      tb += s_t[e] * (aw1[(64 + e) * H_DIM + tid] - aw1[(128 + e) * H_DIM + tid]);
    s_tvec[tid] = tb;
  }
  const float4* ei4 = (const float4*)ei;
  for (int u = tid; u < SP * (E_DIM / 4); u += 256) {
    int s = u >> 4;
    int q = u & 15;
    int ridx = s_sidx[s];
    float4 v = make_float4(0.f, 0.f, 0.f, 0.f);
    if (ridx >= 0) v = ei4[(long)ridx * 16 + q];
    __bf16* dst = &s_beh[s * ROWP + q * 4];
    dst[0] = (__bf16)v.x; dst[1] = (__bf16)v.y;
    dst[2] = (__bf16)v.z; dst[3] = (__bf16)v.w;
  }
  __syncthreads();

  // ---------- phase 1: collapsed attention GEMM via WMMA (K = 64) ----------
  // h^T = W''^T @ beh^T + tvec, with W'' = W1a + W1c + diag(t)*W1d.
  // D tile: M = h-columns (wave-owned 16), N = 16 sequence positions.
  const int col = wv * 16 + li;   // this lane's W'' column (A-matrix row m)

  // A fragments: lane m = col; elems j<8: k = kb+hf*8+j ; j>=8: k = kb+16+hf*8+(j-8)
  v16bf Af[2];
#pragma unroll
  for (int kt = 0; kt < 2; ++kt) {
    const int kb = kt * 32;
#pragma unroll
    for (int j = 0; j < 8; ++j) {
      int k1 = kb + hf * 8 + j;
      int k2 = kb + 16 + hf * 8 + j;
      Af[kt][j]     = (__bf16)(aw1[k1 * H_DIM + col] + aw1[(128 + k1) * H_DIM + col]
                               + s_t[k1] * aw1[(192 + k1) * H_DIM + col]);
      Af[kt][8 + j] = (__bf16)(aw1[k2 * H_DIM + col] + aw1[(128 + k2) * H_DIM + col]
                               + s_t[k2] * aw1[(192 + k2) * H_DIM + col]);
    }
  }

  // per-lane tvec/w2 for its 8 h-columns: m = wv*16 + hf*8 + v
  float tbv[8], w2v[8];
#pragma unroll
  for (int v = 0; v < 8; ++v) {
    int m = wv * 16 + hf * 8 + v;
    tbv[v] = s_tvec[m];
    w2v[v] = aw2[m];
  }

  // 2-tile software pipeline: independent accumulator chains cover WMMA hazards
  for (int st = 0; st < SP / 16; st += 2) {
    const __bf16* r0 = &s_beh[(st * 16 + li) * ROWP];
    const __bf16* r1 = &s_beh[((st + 1) * 16 + li) * ROWP];
    v16bf a0 = b_frag_beh(r0, 0,  hf);
    v16bf a1 = b_frag_beh(r0, 32, hf);
    v16bf c0 = b_frag_beh(r1, 0,  hf);
    v16bf c1 = b_frag_beh(r1, 32, hf);
    v8f acc0 = {0.f, 0.f, 0.f, 0.f, 0.f, 0.f, 0.f, 0.f};
    v8f acc1 = {0.f, 0.f, 0.f, 0.f, 0.f, 0.f, 0.f, 0.f};
    acc0 = WMMA_BF16(Af[0], a0, acc0);
    acc1 = WMMA_BF16(Af[0], c0, acc1);
    acc0 = WMMA_BF16(Af[1], a1, acc0);
    acc1 = WMMA_BF16(Af[1], c1, acc1);

    float x0 = 0.f, x1 = 0.f;
#pragma unroll
    for (int v = 0; v < 8; ++v) {
      x0 += fmaxf(acc0[v] + tbv[v], 0.f) * w2v[v];
      x1 += fmaxf(acc1[v] + tbv[v], 0.f) * w2v[v];
    }
    atomicAdd(&s_scores[st * 16 + li], x0);
    atomicAdd(&s_scores[(st + 1) * 16 + li], x1);
  }
  __syncthreads();

  // ---------- phase 2: softmax over S (att_b2 is a softmax-invariant shift) ----------
  if (tid < 32) {
    float m = -1e30f;
    for (int i = tid; i < S_SEQ; i += 32) m = fmaxf(m, s_scores[i]);
#pragma unroll
    for (int msk = 16; msk >= 1; msk >>= 1) m = fmaxf(m, __shfl_xor(m, msk, 32));
    float dsum = 0.f;
    for (int i = tid; i < S_SEQ; i += 32) {
      float e = __expf(s_scores[i] - m);
      s_scores[i] = e;
      dsum += e;
    }
#pragma unroll
    for (int msk = 16; msk >= 1; msk >>= 1) dsum += __shfl_xor(dsum, msk, 32);
    if (tid == 0) s_inv = 1.0f / dsum;
  }
  __syncthreads();

  // weighted interest: sum_s behavior[s][e] * w[s]
  {
    int e = tid & 63, part = tid >> 6;
    float ws = 0.f;
    int s0 = part * 50;
    for (int s = s0; s < s0 + 50; ++s)
      ws += (float)s_beh[s * ROWP + e] * s_scores[s];
    s_red[tid] = ws;
  }
  __syncthreads();
  if (tid < 64)
    s_comb[128 + tid] = (s_red[tid] + s_red[64 + tid] + s_red[128 + tid] + s_red[192 + tid]) * s_inv;
  __syncthreads();

  // ---------- phase 3: final 256 -> 128 -> 1 MLP ----------
  float val = 0.f;
  if (tid < H_DIM) {
    float accf = fb1[tid];
#pragma unroll 8
    for (int d2 = 0; d2 < 256; ++d2)
      accf = fmaf(s_comb[d2], fw1[d2 * H_DIM + tid], accf);
    val = fmaxf(accf, 0.f) * fw2[tid];
  }
  s_red[tid] = val;
  __syncthreads();
  if (tid < 64) s_red[tid] += s_red[tid + 64];
  __syncthreads();
  if (tid < 32) {
    float x = s_red[tid] + s_red[tid + 32];
#pragma unroll
    for (int msk = 16; msk >= 1; msk >>= 1) x += __shfl_xor(x, msk, 32);
    if (tid == 0) out[b] = x + fb2[0];
  }
  (void)ab2;
}

extern "C" void kernel_launch(void* const* d_in, const int* in_sizes, int n_in,
                              void* d_out, int out_size, void* d_ws, size_t ws_size,
                              hipStream_t stream)
{
  (void)in_sizes; (void)n_in; (void)out_size; (void)d_ws; (void)ws_size;
  din_kernel<<<dim3(B_SZ), dim3(256), 0, stream>>>(
      (const int*)d_in[0], (const int*)d_in[1], (const int*)d_in[2],
      (const float*)d_in[3], (const float*)d_in[4], (const float*)d_in[5],
      (const float*)d_in[6], (const float*)d_in[7],
      (const float*)d_in[8], (const float*)d_in[9],
      (const float*)d_in[10], (const float*)d_in[11],
      (const float*)d_in[12], (const float*)d_in[13],
      (float*)d_out);
}
